// Model_15805479649715
// MI455X (gfx1250) — matrix operand order repaired and checked by reading, not compiled, not run on hardware
//
#include <hip/hip_runtime.h>
#include <hip/hip_bf16.h>

typedef __attribute__((ext_vector_type(16))) _Float16 v16h;
typedef __attribute__((ext_vector_type(8)))  float    v8f;

// Problem constants: B=4, S=127, D=768, H=12, HD=64, NS=128
#define PQ_LD 772   // padded row stride (floats) for P/Q tiles in LDS; 772*4 is 16B-aligned
#define LG_LD 130   // padded row stride for logits/attn tile in LDS

// ---------------- CDNA5 async global->LDS helpers (ASYNCcnt path) ----------------
__device__ __forceinline__ void async_g2l_b128(unsigned lds_off, const void* gaddr) {
    asm volatile("global_load_async_to_lds_b128 %0, %1, off"
                 :: "v"(lds_off), "v"(gaddr) : "memory");
}
__device__ __forceinline__ void wait_async0() {
    asm volatile("s_wait_asynccnt 0x0" ::: "memory");
}
__device__ __forceinline__ unsigned lds_off_of(const void* p) {
    return (unsigned)(uintptr_t)p;   // flat shared ptr: addr[31:0] == LDS offset
}

// ---------------- workspace layout (bytes) ----------------
constexpr size_t OFF_WQA   = 0;                              // 768*12 f32 : Wq folded with aq
constexpr size_t OFF_WKA   = OFF_WQA   + 768*12*4;           // 768*12 f32 : Wk folded with ak
constexpr size_t OFF_W2AF  = OFF_WKA   + 768*12*4;           // 768*12 f32 : eW2 folded with ae
constexpr size_t OFF_BQA   = OFF_W2AF  + 768*12*4;           // 12 f32
constexpr size_t OFF_BKA   = OFF_BQA   + 64;                 // 12 f32
constexpr size_t OFF_EB2A  = OFF_BKA   + 64;                 // 12 f32
constexpr size_t OFF_W2APK = OFF_EB2A  + 64;                 // 24*32*16 f16 : w2a in B-fragment order
constexpr size_t OFF_PP    = OFF_W2APK + 24*32*16*2;         // 508*768 f32 : desc@eW1_top + eb1
constexpr size_t OFF_QQ    = OFF_PP    + 508*768*4;          // 508*768 f32 : desc@eW1_bot
constexpr size_t OFF_VPT   = OFF_QQ    + 508*768*4;          // 768*512 f16 : (nve@Wv+bv)^T, d-major
constexpr size_t OFF_LQ    = OFF_VPT   + 768*512*2;          // 512*12 f32
constexpr size_t OFF_LK    = OFF_LQ    + 512*12*4;           // 512*12 f32
constexpr size_t OFF_GCLS  = OFF_LK    + 512*12*4;           // 508*12 f32 : cls-edge logits

// ---------------- K0a: fold attention vector aW into weight matrices ----------------
__global__ void fold_kernel(const float* __restrict__ Wq, const float* __restrict__ Wk,
                            const float* __restrict__ eW2,
                            const float* __restrict__ bq, const float* __restrict__ bk,
                            const float* __restrict__ eb2, const float* __restrict__ aW,
                            float* __restrict__ Wqa, float* __restrict__ Wka,
                            float* __restrict__ w2aF,
                            float* __restrict__ bqa, float* __restrict__ bka,
                            float* __restrict__ eb2a)
{
    int t = blockIdx.x * blockDim.x + threadIdx.x;
    if (t < 768 * 12) {
        int d = t / 12, h = t % 12;
        float sq = 0.f, sk = 0.f, se = 0.f;
        for (int u = 0; u < 64; ++u) {
            int col = h * 64 + u;
            sq += Wq [d * 768 + col] * aW[u];
            sk += Wk [d * 768 + col] * aW[64 + u];
            se += eW2[d * 768 + col] * aW[128 + u];
        }
        Wqa[t] = sq; Wka[t] = sk; w2aF[t] = se;
    }
    if (t < 12) {
        float sbq = 0.f, sbk = 0.f, sbe = 0.f;
        for (int u = 0; u < 64; ++u) {
            sbq += bq [t * 64 + u] * aW[u];
            sbk += bk [t * 64 + u] * aW[64 + u];
            sbe += eb2[t * 64 + u] * aW[128 + u];
        }
        bqa[t] = sbq; bka[t] = sbk; eb2a[t] = sbe;
    }
}

// ---------------- K0b: pack w2a (768x12, pad N to 16) into per-lane B-fragment order ----------------
__global__ void pack_w2a_kernel(const float* __restrict__ w2aF, _Float16* __restrict__ w2apk)
{
    int g = blockIdx.x * blockDim.x + threadIdx.x; // 0..767 = kt*32 + lane
    if (g >= 24 * 32) return;
    int l  = g & 31;
    int kb = (l & 16) ? 8 : 0;
    int n  = l & 15;
    int k0 = (g >> 5) * 32;
    for (int e = 0; e < 16; ++e) {
        int k = k0 + kb + ((e < 8) ? e : (e + 8));
        w2apk[g * 16 + e] = (n < 12) ? (_Float16)w2aF[k * 12 + n] : (_Float16)0.f;
    }
}

// ---------------- K1: three 512x768x768-class GEMMs with WMMA (P, Q, V^T) ----------------
__global__ __launch_bounds__(256)
void gemm3_kernel(const float* __restrict__ desc, const float* __restrict__ nve,
                  const float* __restrict__ eW1, const float* __restrict__ Wv,
                  const float* __restrict__ eb1, const float* __restrict__ bv,
                  float* __restrict__ Pp, float* __restrict__ Qq,
                  _Float16* __restrict__ VpT)
{
    const int z    = blockIdx.z;                    // 0: P, 1: Q, 2: V^T
    const int lane = threadIdx.x & 31;
    const int wv   = threadIdx.x >> 5;
    const int nt   = blockIdx.x * 8 + wv;           // 0..47  (N tile)
    const int mt   = blockIdx.y;                    // 0..31  (M tile)
    const int n    = nt * 16 + (lane & 15);
    const int M    = (z < 2) ? 508 : 512;
    const float* A = (z < 2) ? desc : nve;
    const float* W = (z == 0) ? eW1 : (z == 1) ? (eW1 + 768 * 768) : Wv;

    const int khalf = (lane & 16) >> 1;             // 0 or 8
    const int kb16  = lane & 16;                    // 0 or 16
    const int mrowA = mt * 16 + (lane & 15);
    const int mA    = (mrowA < M) ? mrowA : 0;

    v8f c = {};
    for (int kt = 0; kt < 24; ++kt) {
        const int k0 = kt * 32;
        v16h a;
        const float* Ar = A + (size_t)mA * 768 + k0 + kb16;
        #pragma unroll
        for (int e = 0; e < 16; ++e) {
            int kk = e;
            a[e] = (_Float16)Ar[kk];
        }
        v16h bf;
        const float* Wp = W + (size_t)(k0 + kb16) * 768 + n;
        #pragma unroll
        for (int e = 0; e < 16; ++e) bf[e] = (_Float16)Wp[e * 768];
        c = __builtin_amdgcn_wmma_f32_16x16x32_f16(false, a, false, bf, (short)0, c, false, false);
    }
    if (z == 2) {
        const float bias = bv[n];
        #pragma unroll
        for (int v = 0; v < 8; ++v) {
            int m = mt * 16 + v + ((lane & 16) ? 8 : 0);      // m < 512 always
            VpT[(size_t)n * 512 + m] = (_Float16)(c[v] + bias);
        }
    } else {
        float* out = (z == 0) ? Pp : Qq;
        const float bias = (z == 0) ? eb1[n] : 0.f;
        #pragma unroll
        for (int v = 0; v < 8; ++v) {
            int m = mt * 16 + v + ((lane & 16) ? 8 : 0);
            if (m < M) out[(size_t)m * 768 + n] = c[v] + bias;
        }
    }
}

// ---------------- K2: per-node folded q/k logits: lq[b,n,h], lk[b,n,h] ----------------
__global__ __launch_bounds__(128)
void qk_logit_kernel(const float* __restrict__ nve, const float* __restrict__ Wqa,
                     const float* __restrict__ Wka, const float* __restrict__ bqa,
                     const float* __restrict__ bka,
                     float* __restrict__ lq, float* __restrict__ lk)
{
    const int lane = threadIdx.x & 31;
    const int idx  = blockIdx.x * 4 + (threadIdx.x >> 5);  // 0..511
    float aq[12] = {}, ak[12] = {};
    const float* x = nve + (size_t)idx * 768;
    for (int d = lane; d < 768; d += 32) {
        float xv = x[d];
        const float* wq = Wqa + d * 12;
        const float* wk = Wka + d * 12;
        #pragma unroll
        for (int h = 0; h < 12; ++h) { aq[h] += xv * wq[h]; ak[h] += xv * wk[h]; }
    }
    #pragma unroll
    for (int h = 0; h < 12; ++h)
        for (int off = 16; off > 0; off >>= 1) {
            aq[h] += __shfl_xor(aq[h], off, 32);
            ak[h] += __shfl_xor(ak[h], off, 32);
        }
    if (lane == 0) {
        #pragma unroll
        for (int h = 0; h < 12; ++h) {
            lq[idx * 12 + h] = aq[h] + bqa[h];
            lk[idx * 12 + h] = ak[h] + bka[h];
        }
    }
}

// ---------------- K2b: cls-edge logits g[b,s,h] = relu(LN(P_s+Q_s)) . w2a + eb2a ----------------
__global__ __launch_bounds__(128)
void cls_logit_kernel(const float* __restrict__ Pp, const float* __restrict__ Qq,
                      const float* __restrict__ ln_g, const float* __restrict__ ln_b,
                      const float* __restrict__ w2aF, const float* __restrict__ eb2a,
                      float* __restrict__ gcls)
{
    const int lane = threadIdx.x & 31;
    const int idx  = blockIdx.x * 4 + (threadIdx.x >> 5);  // 0..507
    if (idx >= 508) return;
    const float* P = Pp + (size_t)idx * 768;
    const float* Q = Qq + (size_t)idx * 768;
    float s1 = 0.f, s2 = 0.f;
    for (int d = lane; d < 768; d += 32) { float x = P[d] + Q[d]; s1 += x; s2 += x * x; }
    for (int off = 16; off > 0; off >>= 1) { s1 += __shfl_xor(s1, off, 32); s2 += __shfl_xor(s2, off, 32); }
    const float mean = s1 * (1.f / 768.f);
    const float var  = s2 * (1.f / 768.f) - mean * mean;
    const float rstd = rsqrtf(var + 1e-5f);
    float acc[12] = {};
    for (int d = lane; d < 768; d += 32) {
        float x  = P[d] + Q[d];
        float xn = fmaxf((x - mean) * rstd * ln_g[d] + ln_b[d], 0.f);
        const float* wp = w2aF + d * 12;
        #pragma unroll
        for (int h = 0; h < 12; ++h) acc[h] += xn * wp[h];
    }
    #pragma unroll
    for (int h = 0; h < 12; ++h)
        for (int off = 16; off > 0; off >>= 1) acc[h] += __shfl_xor(acc[h], off, 32);
    if (lane == 0)
        for (int h = 0; h < 12; ++h) gcls[idx * 12 + h] = acc[h] + eb2a[h];
}

// ---------------- K3: fused edge-LN-MLP + logits + softmax + context ----------------
// Grid: (8 i-tiles, B). Block: 256 (8 waves). Dynamic LDS ~200 KB (fits 320 KB WGP LDS).
__global__ __launch_bounds__(256)
void edge_attn_kernel(const float* __restrict__ Pp, const float* __restrict__ Qq,
                      const _Float16* __restrict__ VpT,
                      const float* __restrict__ lq, const float* __restrict__ lk,
                      const float* __restrict__ gcls,
                      const _Float16* __restrict__ w2apk,
                      const float* __restrict__ eb2a,
                      const float* __restrict__ ln_g, const float* __restrict__ ln_b,
                      const float* __restrict__ mask_M, const float* __restrict__ ab,
                      float* __restrict__ out_basis, float* __restrict__ out_attn)
{
    extern __shared__ char smem_raw[];
    float* Psm = (float*)smem_raw;            // 16*PQ_LD
    float* Qsm = Psm + 16 * PQ_LD;            // 16*PQ_LD
    float* gsm = Qsm + 16 * PQ_LD;            // 768
    float* bsm = gsm + 768;                   // 768
    float* Lsm = bsm + 768;                   // 12*16*LG_LD (logits -> attn)

    const int b    = blockIdx.y;
    const int i0   = blockIdx.x * 16;
    const int tid  = threadIdx.x;
    const int lane = tid & 31;
    const int w    = tid >> 5;
    const float abv = ab[0];

    const int khalf = (lane & 16) >> 1;  // 0 or 8
    const int kb16  = lane & 16;         // 0 or 16
    const int n15   = lane & 15;

    // Async-stage the 16 P-rows of this i-tile + LN params into LDS (ASYNCcnt path).
    {
        const unsigned p_off = lds_off_of(Psm);
        for (int idx = tid; idx < 16 * 192; idx += 256) {   // 192 x 16B chunks per 768-f32 row
            int r = idx / 192, ck = idx % 192;
            int i = i0 + r;
            int s = (i >= 1) ? (i - 1) : 0;                 // i==0 row is overwritten later
            async_g2l_b128(p_off + (unsigned)(r * PQ_LD + ck * 4) * 4,
                           Pp + ((size_t)b * 127 + s) * 768 + ck * 4);
        }
        const unsigned g_off = lds_off_of(gsm);
        const unsigned b_off = lds_off_of(bsm);
        for (int ck = tid; ck < 192; ck += 256) {
            async_g2l_b128(g_off + (unsigned)ck * 16, ln_g + ck * 4);
            async_g2l_b128(b_off + (unsigned)ck * 16, ln_b + ck * 4);
        }
        wait_async0();
    }

    for (int jt = 0; jt < 8; ++jt) {
        __syncthreads();
        {   // async-stage the 16 Q-rows of this j-tile
            const unsigned q_off = lds_off_of(Qsm);
            for (int idx = tid; idx < 16 * 192; idx += 256) {
                int r = idx / 192, ck = idx % 192;
                int j = jt * 16 + r;
                int s = (j >= 1) ? (j - 1) : 0;             // j==0 column masked anyway
                async_g2l_b128(q_off + (unsigned)(r * PQ_LD + ck * 4) * 4,
                               Qq + ((size_t)b * 127 + s) * 768 + ck * 4);
            }
            wait_async0();
        }
        __syncthreads();

        for (int rr = 0; rr < 2; ++rr) {
            const int r = w + rr * 8;                  // i row within tile
            const int i = i0 + r;

            // prefetch the mask row this (i, h) will need in the epilogue
            if (i >= 1 && n15 < 12)
                __builtin_prefetch(mask_M + (((size_t)b * 12 + n15) * 127 + (i - 1)) * 127 + jt * 16, 0, 0);

            // pass 1: per-edge LN statistics (each lane-pair owns one edge)
            float s1 = 0.f, s2 = 0.f;
            {
                const float* Pr = Psm + r * PQ_LD;
                const float* Qr = Qsm + n15 * PQ_LD;
                const int d0 = (lane >> 4) * 384;
                for (int d = d0; d < d0 + 384; ++d) {
                    float x = Pr[d] + Qr[d];
                    s1 += x; s2 += x * x;
                }
            }
            s1 += __shfl_xor(s1, 16, 32);
            s2 += __shfl_xor(s2, 16, 32);
            const float mean = s1 * (1.f / 768.f);
            const float var  = s2 * (1.f / 768.f) - mean * mean;
            const float rstd = rsqrtf(var + 1e-5f);

            // pass 2: build relu(LN(x)) A-fragments on the fly, WMMA against w2a
            v8f c = {};
            for (int kt = 0; kt < 24; ++kt) {
                const int k0 = kt * 32;
                v16h a;
                const float* Pr  = Psm + r   * PQ_LD + k0 + khalf;
                const float* Qr  = Qsm + n15 * PQ_LD + k0 + khalf;
                const float* gg  = gsm + k0 + khalf;
                const float* bb2 = bsm + k0 + khalf;
                #pragma unroll
                for (int e = 0; e < 16; ++e) {
                    int kk = (e < 8) ? e : (e + 8);
                    float x  = Pr[kk] + Qr[kk];
                    float xn = (x - mean) * rstd * gg[kk] + bb2[kk];
                    a[e] = (_Float16)fmaxf(xn, 0.f);
                }
                v16h bf = *(const v16h*)(w2apk + (size_t)(k0 + lane) * 16);
                c = __builtin_amdgcn_wmma_f32_16x16x32_f16(false, a, false, bf, (short)0, c, false, false);
            }

            // assemble logits: C rows = j edges, cols = heads
            const int h = n15;
            if (h < 12) {
                const float lqv = lq[((size_t)b * 128 + i) * 12 + h];
                const float e2  = eb2a[h];
                #pragma unroll
                for (int v = 0; v < 8; ++v) {
                    int jl = v + ((lane & 16) ? 8 : 0);
                    int j  = jt * 16 + jl;
                    float val;
                    if (j == 0 || j == i) {
                        val = -1e9f;                               // adj==0 (col 0 & diagonal)
                    } else if (i == 0) {
                        val = lqv + lk[((size_t)b * 128 + j) * 12 + h] + abv
                            + gcls[((size_t)b * 127 + (j - 1)) * 12 + h];
                    } else {
                        float p = mask_M[(((size_t)b * 12 + h) * 127 + (i - 1)) * 127 + (j - 1)];
                        p = fminf(fmaxf(p, 1e-6f), 1.f - 1e-6f);
                        float bias = __logf(p) - __logf(1.f - p);  // * GAMMA(=1)
                        val = c[v] + e2 + abv + lqv
                            + lk[((size_t)b * 128 + j) * 12 + h] + bias;
                    }
                    Lsm[(h * 16 + r) * LG_LD + j] = val;
                }
            }
        }
    }
    __syncthreads();

    // softmax over j (one thread per (h, i-row)); write attn, keep normalized in LDS
    if (tid < 192) {
        const int h = tid / 16;
        const int r = tid % 16;
        const int i = i0 + r;
        float* row = Lsm + (h * 16 + r) * LG_LD;
        float m = -3.4e38f;
        for (int j = 0; j < 128; ++j) m = fmaxf(m, row[j]);
        float s = 0.f;
        for (int j = 0; j < 128; ++j) { float e = __expf(row[j] - m); row[j] = e; s += e; }
        const float inv = 1.f / s;
        float* ar = out_attn + (((size_t)b * 12 + h) * 128 + i) * 128;
        for (int j = 0; j < 128; ++j) { float p = row[j] * inv; row[j] = p; ar[j] = p; }
    }
    __syncthreads();

    // context = attn @ V per head, WMMA 16(i) x 16(d) x 128(j); V^T gives j-contiguous B-fragments
    for (int it = 0; it < 6; ++it) {
        const int job = w * 6 + it;     // 48 jobs = 12 heads * 4 d-tiles
        const int h  = job >> 2;
        const int dt = job & 3;
        const _Float16* vrow = VpT + (size_t)(h * 64 + dt * 16 + n15) * 512 + b * 128;
        v8f c = {};
        for (int kt = 0; kt < 4; ++kt) {
            const int k0 = kt * 32;
            v16h a;
            const float* ar = Lsm + (h * 16 + n15) * LG_LD + k0 + kb16;
            #pragma unroll
            for (int e = 0; e < 16; ++e) {
                int kk = e;
                a[e] = (_Float16)ar[kk];
            }
            v16h bf = *(const v16h*)(vrow + k0 + kb16);
            c = __builtin_amdgcn_wmma_f32_16x16x32_f16(false, a, false, bf, (short)0, c, false, false);
        }
        #pragma unroll
        for (int v = 0; v < 8; ++v) {
            int m = v + ((lane & 16) ? 8 : 0);
            out_basis[(((size_t)b * 128 + i0 + m) * 12 + h) * 64 + dt * 16 + n15] = c[v];
        }
    }
}

// ---------------- host-side launcher ----------------
extern "C" void kernel_launch(void* const* d_in, const int* in_sizes, int n_in,
                              void* d_out, int out_size, void* d_ws, size_t ws_size,
                              hipStream_t stream) {
    const float* desc   = (const float*)d_in[0];
    const float* nve    = (const float*)d_in[1];
    const float* mask_M = (const float*)d_in[2];
    const float* Wq     = (const float*)d_in[3];
    const float* bq     = (const float*)d_in[4];
    const float* Wk     = (const float*)d_in[5];
    const float* bk     = (const float*)d_in[6];
    const float* Wv     = (const float*)d_in[7];
    const float* bv     = (const float*)d_in[8];
    const float* eW1    = (const float*)d_in[9];
    const float* eb1    = (const float*)d_in[10];
    const float* ln_g   = (const float*)d_in[11];
    const float* ln_b   = (const float*)d_in[12];
    const float* eW2    = (const float*)d_in[13];
    const float* eb2    = (const float*)d_in[14];
    const float* aW     = (const float*)d_in[15];
    const float* ab     = (const float*)d_in[16];

    char* ws = (char*)d_ws;
    float*     Wqa   = (float*)(ws + OFF_WQA);
    float*     Wka   = (float*)(ws + OFF_WKA);
    float*     w2aF  = (float*)(ws + OFF_W2AF);
    float*     bqa   = (float*)(ws + OFF_BQA);
    float*     bka   = (float*)(ws + OFF_BKA);
    float*     eb2a  = (float*)(ws + OFF_EB2A);
    _Float16*  w2apk = (_Float16*)(ws + OFF_W2APK);
    float*     Pp    = (float*)(ws + OFF_PP);
    float*     Qq    = (float*)(ws + OFF_QQ);
    _Float16*  VpT   = (_Float16*)(ws + OFF_VPT);
    float*     lq    = (float*)(ws + OFF_LQ);
    float*     lk    = (float*)(ws + OFF_LK);
    float*     gcls  = (float*)(ws + OFF_GCLS);

    float* out_basis = (float*)d_out;                    // (B,NS,H,HD) = 393216 f32
    float* out_attn  = out_basis + 4 * 128 * 12 * 64;    // (B,H,NS,NS) = 786432 f32

    fold_kernel<<<36, 256, 0, stream>>>(Wq, Wk, eW2, bq, bk, eb2, aW,
                                        Wqa, Wka, w2aF, bqa, bka, eb2a);
    pack_w2a_kernel<<<3, 256, 0, stream>>>(w2aF, w2apk);
    gemm3_kernel<<<dim3(6, 32, 3), 256, 0, stream>>>(desc, nve, eW1, Wv, eb1, bv, Pp, Qq, VpT);
    qk_logit_kernel<<<128, 128, 0, stream>>>(nve, Wqa, Wka, bqa, bka, lq, lk);
    cls_logit_kernel<<<127, 128, 0, stream>>>(Pp, Qq, ln_g, ln_b, w2aF, eb2a, gcls);

    size_t shmem = (size_t)(2 * 16 * PQ_LD + 2 * 768 + 12 * 16 * LG_LD) * sizeof(float);
    edge_attn_kernel<<<dim3(8, 4), 256, shmem, stream>>>(Pp, Qq, VpT, lq, lk, gcls, w2apk,
                                                         eb2a, ln_g, ln_b, mask_M, ab,
                                                         out_basis, out_attn);
}
